// Seq2SeqAttention_78254304133447
// MI455X (gfx1250) — compile-verified, matrix operand order checked by reading, not run on hardware
//
#include <hip/hip_runtime.h>
#include <hip/hip_bf16.h>

#define DEV __device__ __forceinline__

typedef __attribute__((ext_vector_type(16))) _Float16 v16h;
typedef __attribute__((ext_vector_type(8)))  _Float16 v8h;
typedef __attribute__((ext_vector_type(8)))  float    v8f;

union AV { v16h v; v8h h[2]; };

constexpr int EMB    = 512;
constexpr int HID    = 1024;
constexpr int G4     = 4 * HID;        // 4096
constexpr int BSZ    = 16;
constexpr int LE     = 256;
constexpr int LD     = 64;
constexpr int VOC    = 32000;
constexpr int ROWS_E = BSZ * LE;       // 4096
constexpr int ROWS_D = BSZ * LD;       // 1024
constexpr int DEC_IN = EMB + HID;      // 1536
constexpr int CAT2   = 2 * HID;        // 2048

DEV float sigm(float x) { return 1.0f / (1.0f + __expf(-x)); }

// Lane coordinates for 16x16x32 f16 WMMA (wave32), per ISA 7.12.2.
DEV void lane_coords(int& l15, int& kb, int& mh) {
  int lane = threadIdx.x & 31;
  l15 = lane & 15;
  kb  = (lane & 16) >> 1;   // 0 or 8
  mh  = (lane >> 4) & 1;    // C/D row-half
}

// MB x NB register-blocked tile GEMM, compile-time K, grouped loads.
// Low register pressure; latency hidden by multi-wave occupancy.
template<int MB, int NB, int K>
DEV void wmma_mn(const _Float16* __restrict__ ap, size_t astride,
                 const _Float16* __restrict__ wp, size_t wstride,
                 v8f (&acc)[MB][NB]) {
#pragma unroll
  for (int k = 0; k < K; k += 32) {
    AV a[MB], b[NB];
#pragma unroll
    for (int m = 0; m < MB; ++m) {
      a[m].h[0] = *(const v8h*)(ap + (size_t)m * astride + k);
      a[m].h[1] = *(const v8h*)(ap + (size_t)m * astride + k + 16);
    }
#pragma unroll
    for (int t = 0; t < NB; ++t) {
      const _Float16* w = wp + (size_t)t * wstride + k;
      b[t].h[0] = *(const v8h*)(w);
      b[t].h[1] = *(const v8h*)(w + 16);
    }
#pragma unroll
    for (int t = 0; t < NB; ++t)
#pragma unroll
      for (int m = 0; m < MB; ++m)
        acc[m][t] = __builtin_amdgcn_wmma_f32_16x16x32_f16(false, a[m].v, false, b[t].v,
                                                           (short)0, acc[m][t],
                                                           false, false);
  }
}

// Explicitly 2-stage software-pipelined variant for single-wave kernels:
// the k+32 fragments are loaded (by dataflow) before the k-step WMMAs issue,
// so WMMA waits see at most the previous clause outstanding. Needs the
// waves_per_eu(1,1) pressure budget (~130 VGPRs live).
template<int NB, int K>
DEV void wmma_nb_pipe(const _Float16* __restrict__ ap,
                      const _Float16* __restrict__ wp, size_t wstride,
                      v8f (&acc)[NB]) {
  AV a[2];
  AV b[2][NB];
  a[0].h[0] = *(const v8h*)(ap);
  a[0].h[1] = *(const v8h*)(ap + 16);
#pragma unroll
  for (int t = 0; t < NB; ++t) {
    const _Float16* w = wp + (size_t)t * wstride;
    b[0][t].h[0] = *(const v8h*)(w);
    b[0][t].h[1] = *(const v8h*)(w + 16);
  }
#pragma unroll
  for (int k = 0; k < K; k += 32) {
    const int cur = (k >> 5) & 1;
    const int nxt = cur ^ 1;
    if (k + 32 < K) {
      a[nxt].h[0] = *(const v8h*)(ap + k + 32);
      a[nxt].h[1] = *(const v8h*)(ap + k + 48);
#pragma unroll
      for (int t = 0; t < NB; ++t) {
        const _Float16* w = wp + (size_t)t * wstride + k + 32;
        b[nxt][t].h[0] = *(const v8h*)(w);
        b[nxt][t].h[1] = *(const v8h*)(w + 16);
      }
    }
#pragma unroll
    for (int t = 0; t < NB; ++t)
      acc[t] = __builtin_amdgcn_wmma_f32_16x16x32_f16(false, a[cur].v, false, b[cur][t].v,
                                                      (short)0, acc[t], false, false);
  }
}

// ---------------- utility kernels ----------------

__global__ void f32_to_f16(const float* __restrict__ s, _Float16* __restrict__ d, int n) {
  int i = blockIdx.x * blockDim.x + threadIdx.x;
  int stride = gridDim.x * blockDim.x;
  for (; i < n; i += stride) d[i] = (_Float16)s[i];
}

__global__ void gather_embed(const int* __restrict__ tok, const float* __restrict__ emb,
                             _Float16* __restrict__ dst, int nrows) {
  int row = blockIdx.x;
  if (row >= nrows) return;
  const float* src = emb + (size_t)tok[row] * EMB;
  for (int e = threadIdx.x; e < EMB; e += blockDim.x)
    dst[(size_t)row * EMB + e] = (_Float16)src[e];
}

// ------- generic batched GEMM: D[M,N] = A[M,K] @ W[N,K]^T + b1 + b2 -------
// One wave per block; each wave owns a 32(M) x 64(N) strip: 2x4 WMMA tiles,
// 1.5 b128 loads per WMMA, 8 independent accumulation chains.
template<int K>
__global__ void gemm_bias_f32(const _Float16* __restrict__ A, int lda,
                              const _Float16* __restrict__ W, int ldw,
                              const float* __restrict__ bias1,
                              const float* __restrict__ bias2,
                              float* __restrict__ D, int ldo) {
  int n0 = blockIdx.x * 64;
  int m0 = blockIdx.y * 32;
  int l15, kb, mh; lane_coords(l15, kb, mh);
  const _Float16* ap = A + (size_t)(m0 + l15) * lda + kb;
  const _Float16* wp = W + (size_t)(n0 + l15) * ldw + kb;
  __builtin_prefetch(wp, 0, 3);
  v8f acc[2][4] = {};
  wmma_mn<2, 4, K>(ap, (size_t)16 * lda, wp, (size_t)16 * ldw, acc);
#pragma unroll
  for (int t = 0; t < 4; ++t) {
    int n = n0 + t * 16 + l15;
    float bv = (bias1 ? bias1[n] : 0.0f) + (bias2 ? bias2[n] : 0.0f);
#pragma unroll
    for (int m = 0; m < 2; ++m) {
      int mb = m0 + m * 16 + mh * 8;
#pragma unroll
      for (int r = 0; r < 8; ++r)
        D[(size_t)(mb + r) * ldo + n] = acc[m][t][r] + bv;
    }
  }
}

// ---------------- encoder LSTM step ----------------
// grid: 64 blocks (hidden tiles of 16) x 1 wave. Only ~64 waves chip-wide ->
// pin occupancy to 1 wave/EU so the full VGPR file backs the 2-deep pipeline.
// The 4 gates are the N-block: gate g rows at offset g*HID -> wstride=HID*HID.
__global__ void __launch_bounds__(32)
__attribute__((amdgpu_waves_per_eu(1, 1)))
enc_step(const float* __restrict__ xpre,
         const _Float16* __restrict__ hin,
         const _Float16* __restrict__ Whh,
         float* __restrict__ c,
         float* __restrict__ hout_f32,
         _Float16* __restrict__ hout_f16,
         float* __restrict__ enc_out,
         int t) {
  int j = blockIdx.x;
  int l15, kb, mh; lane_coords(l15, kb, mh);
  const _Float16* ap = hin + (size_t)l15 * HID + kb;
  const _Float16* wp = Whh + (size_t)(j * 16 + l15) * HID + kb;
  v8f acc[4] = {};
  wmma_nb_pipe<4, HID>(ap, wp, (size_t)HID * HID, acc);
  int nh = j * 16 + l15;
  int mb = mh * 8;
#pragma unroll
  for (int r = 0; r < 8; ++r) {
    int b = mb + r;
    size_t xrow = (size_t)(b * LE + t) * G4;
    float pi = acc[0][r] + xpre[xrow + 0 * HID + nh];
    float pf = acc[1][r] + xpre[xrow + 1 * HID + nh];
    float pg = acc[2][r] + xpre[xrow + 2 * HID + nh];
    float po = acc[3][r] + xpre[xrow + 3 * HID + nh];
    float cv = c[b * HID + nh];
    float cn = sigm(pf) * cv + sigm(pi) * tanhf(pg);
    float hn = sigm(po) * tanhf(cn);
    c[b * HID + nh] = cn;
    hout_f32[b * HID + nh] = hn;
    hout_f16[b * HID + nh] = (_Float16)hn;
    enc_out[((size_t)b * LE + t) * HID + nh] = hn;
  }
}

// ---------------- decoder attention ----------------
// grid: BSZ blocks, 256 threads. h[b] (4KB) is staged into LDS with async
// global->LDS copies (ASYNCcnt path), then reused 256x by the score dots.
__global__ void attn_step(const float* __restrict__ enc_out,
                          const float* __restrict__ h,
                          const _Float16* __restrict__ demb,  // [B*LD, EMB], row = b*LD+t
                          _Float16* __restrict__ cellin,      // [B, DEC_IN]
                          _Float16* __restrict__ cat2,        // [B, CAT2], writes cols HID..
                          int t) {
  __shared__ float hlds[HID];
  __shared__ float sc[LE];
  __shared__ float red[256];
  int b = blockIdx.x;
  int tid = threadIdx.x;
  const float* eo = enc_out + (size_t)b * LE * HID;
  const float* hb = h + (size_t)b * HID;

  // async stage h[b] -> LDS: 256 threads x 16B = 4KB
  {
    unsigned int lds = (unsigned int)(size_t)&hlds[tid * 4];
    unsigned long long ga = (unsigned long long)(size_t)(hb + tid * 4);
    asm volatile("global_load_async_to_lds_b128 %0, %1, off"
                 :: "v"(lds), "v"(ga) : "memory");
    asm volatile("s_wait_asynccnt 0x0" ::: "memory");
  }
  __syncthreads();

  // dot-attention scores
  {
    const float* row = eo + (size_t)tid * HID;
    float s = 0.0f;
    for (int k = 0; k < HID; ++k) s += row[k] * hlds[k];
    sc[tid] = s;
  }
  __syncthreads();
  red[tid] = sc[tid];
  __syncthreads();
  for (int s = 128; s > 0; s >>= 1) {
    if (tid < s) red[tid] = fmaxf(red[tid], red[tid + s]);
    __syncthreads();
  }
  float mx = red[0];
  __syncthreads();
  float e = __expf(sc[tid] - mx);
  red[tid] = e;
  __syncthreads();
  for (int s = 128; s > 0; s >>= 1) {
    if (tid < s) red[tid] += red[tid + s];
    __syncthreads();
  }
  float inv = 1.0f / red[0];
  __syncthreads();
  sc[tid] = e * inv;
  __syncthreads();
  // context = aw @ enc_out
  for (int hh = tid; hh < HID; hh += blockDim.x) {
    float ctx = 0.0f;
    for (int l = 0; l < LE; ++l) ctx += sc[l] * eo[(size_t)l * HID + hh];
    _Float16 ch = (_Float16)ctx;
    cellin[(size_t)b * DEC_IN + EMB + hh] = ch;
    cat2[(size_t)b * CAT2 + HID + hh] = ch;
  }
  // cell-input embedding slice
  const _Float16* er = demb + (size_t)(b * LD + t) * EMB;
  for (int e2 = tid; e2 < EMB; e2 += blockDim.x)
    cellin[(size_t)b * DEC_IN + e2] = er[e2];
}

// ---------------- decoder LSTM step ----------------
__global__ void __launch_bounds__(32)
__attribute__((amdgpu_waves_per_eu(1, 1)))
dec_step(const _Float16* __restrict__ cellin,
         const _Float16* __restrict__ hin,
         const _Float16* __restrict__ Wih,   // [G4, DEC_IN]
         const _Float16* __restrict__ Whh,   // [G4, HID]
         const float* __restrict__ bih,
         const float* __restrict__ bhh,
         float* __restrict__ c,
         float* __restrict__ hout_f32,
         _Float16* __restrict__ hout_f16,
         _Float16* __restrict__ cat2) {      // writes cols 0..HID-1
  int j = blockIdx.x;
  int l15, kb, mh; lane_coords(l15, kb, mh);
  const _Float16* ap_x = cellin + (size_t)l15 * DEC_IN + kb;
  const _Float16* ap_h = hin + (size_t)l15 * HID + kb;
  const _Float16* wp_x = Wih + (size_t)(j * 16 + l15) * DEC_IN + kb;
  const _Float16* wp_h = Whh + (size_t)(j * 16 + l15) * HID + kb;
  v8f acc[4] = {};
  wmma_nb_pipe<4, DEC_IN>(ap_x, wp_x, (size_t)HID * DEC_IN, acc);
  wmma_nb_pipe<4, HID>(ap_h, wp_h, (size_t)HID * HID, acc);
  int nh = j * 16 + l15;
  int mb = mh * 8;
#pragma unroll
  for (int r = 0; r < 8; ++r) {
    int b = mb + r;
    float pi = acc[0][r] + bih[0 * HID + nh] + bhh[0 * HID + nh];
    float pf = acc[1][r] + bih[1 * HID + nh] + bhh[1 * HID + nh];
    float pg = acc[2][r] + bih[2 * HID + nh] + bhh[2 * HID + nh];
    float po = acc[3][r] + bih[3 * HID + nh] + bhh[3 * HID + nh];
    float cv = c[b * HID + nh];
    float cn = sigm(pf) * cv + sigm(pi) * tanhf(pg);
    float hn = sigm(po) * tanhf(cn);
    c[b * HID + nh] = cn;
    hout_f32[b * HID + nh] = hn;
    hout_f16[b * HID + nh] = (_Float16)hn;
    cat2[(size_t)b * CAT2 + nh] = (_Float16)hn;
  }
}

// ------- attentional projection: proj = tanh([h,ctx] @ W_attn^T + b) -------
// grid: HID/64 = 16 blocks x 1 wave, N-blocked x4, pipelined.
__global__ void __launch_bounds__(32)
__attribute__((amdgpu_waves_per_eu(1, 1)))
proj_step(const _Float16* __restrict__ cat2,
          const _Float16* __restrict__ Wattn,  // [HID, CAT2]
          const float* __restrict__ battn,
          _Float16* __restrict__ proj,         // [B*LD, HID], row = b*LD+t
          int t) {
  int n0 = blockIdx.x * 64;
  int l15, kb, mh; lane_coords(l15, kb, mh);
  const _Float16* ap = cat2 + (size_t)l15 * CAT2 + kb;
  const _Float16* wp = Wattn + (size_t)(n0 + l15) * CAT2 + kb;
  v8f acc[4] = {};
  wmma_nb_pipe<4, CAT2>(ap, wp, (size_t)16 * CAT2, acc);
  int mb = mh * 8;
#pragma unroll
  for (int tt = 0; tt < 4; ++tt) {
    int nh = n0 + tt * 16 + l15;
    float bv = battn[nh];
#pragma unroll
    for (int r = 0; r < 8; ++r) {
      int b = mb + r;
      proj[((size_t)b * LD + t) * HID + nh] = (_Float16)tanhf(acc[tt][r] + bv);
    }
  }
}

// ---------------- host driver ----------------

extern "C" void kernel_launch(void* const* d_in, const int* in_sizes, int n_in,
                              void* d_out, int out_size, void* d_ws, size_t ws_size,
                              hipStream_t stream) {
  (void)in_sizes; (void)n_in; (void)out_size; (void)ws_size;
  const int*   enc_input = (const int*)d_in[0];
  const int*   dec_input = (const int*)d_in[1];
  const float* enc_emb   = (const float*)d_in[2];
  const float* dec_emb   = (const float*)d_in[3];
  const float* W_ih_enc  = (const float*)d_in[4];
  const float* W_hh_enc  = (const float*)d_in[5];
  const float* b_ih_enc  = (const float*)d_in[6];
  const float* b_hh_enc  = (const float*)d_in[7];
  const float* W_ih_dec  = (const float*)d_in[8];
  const float* W_hh_dec  = (const float*)d_in[9];
  const float* b_ih_dec  = (const float*)d_in[10];
  const float* b_hh_dec  = (const float*)d_in[11];
  const float* W_attn    = (const float*)d_in[12];
  const float* b_attn    = (const float*)d_in[13];
  const float* W_out     = (const float*)d_in[14];
  const float* b_out     = (const float*)d_in[15];
  float* out = (float*)d_out;

  char* ws = (char*)d_ws;
  size_t off = 0;
  auto alloc = [&](size_t bytes) -> void* {
    off = (off + 255) & ~size_t(255);
    void* p = ws + off;
    off += bytes;
    return p;
  };

  _Float16* Wih_e = (_Float16*)alloc((size_t)G4 * EMB * 2);
  _Float16* Whh_e = (_Float16*)alloc((size_t)G4 * HID * 2);
  _Float16* Wih_d = (_Float16*)alloc((size_t)G4 * DEC_IN * 2);
  _Float16* Whh_d = (_Float16*)alloc((size_t)G4 * HID * 2);
  _Float16* Wat   = (_Float16*)alloc((size_t)HID * CAT2 * 2);
  _Float16* Wo    = (_Float16*)alloc((size_t)VOC * HID * 2);
  _Float16* xenc  = (_Float16*)alloc((size_t)ROWS_E * EMB * 2);
  _Float16* demb  = (_Float16*)alloc((size_t)ROWS_D * EMB * 2);
  float*    xpre  = (float*)   alloc((size_t)ROWS_E * G4 * 4);
  float*    encout= (float*)   alloc((size_t)BSZ * LE * HID * 4);
  float*    h32   = (float*)   alloc((size_t)BSZ * HID * 4);
  float*    cbuf  = (float*)   alloc((size_t)BSZ * HID * 4);
  _Float16* hbuf0 = (_Float16*)alloc((size_t)BSZ * HID * 2);
  _Float16* hbuf1 = (_Float16*)alloc((size_t)BSZ * HID * 2);
  _Float16* cellin= (_Float16*)alloc((size_t)BSZ * DEC_IN * 2);
  _Float16* cat2  = (_Float16*)alloc((size_t)BSZ * CAT2 * 2);
  _Float16* proj  = (_Float16*)alloc((size_t)ROWS_D * HID * 2);
  _Float16* hbuf[2] = {hbuf0, hbuf1};

  // 1) weights -> f16
  f32_to_f16<<<1024, 256, 0, stream>>>(W_ih_enc, Wih_e, G4 * EMB);
  f32_to_f16<<<1024, 256, 0, stream>>>(W_hh_enc, Whh_e, G4 * HID);
  f32_to_f16<<<1024, 256, 0, stream>>>(W_ih_dec, Wih_d, G4 * DEC_IN);
  f32_to_f16<<<1024, 256, 0, stream>>>(W_hh_dec, Whh_d, G4 * HID);
  f32_to_f16<<<1024, 256, 0, stream>>>(W_attn,   Wat,   HID * CAT2);
  f32_to_f16<<<4096, 256, 0, stream>>>(W_out,    Wo,    VOC * HID);

  // 2) embedding gathers (padding row 0 already zero in the tables)
  gather_embed<<<ROWS_E, 256, 0, stream>>>(enc_input, enc_emb, xenc, ROWS_E);
  gather_embed<<<ROWS_D, 256, 0, stream>>>(dec_input, dec_emb, demb, ROWS_D);

  // 3) zero recurrent state
  hipMemsetAsync(h32,   0, (size_t)BSZ * HID * 4, stream);
  hipMemsetAsync(cbuf,  0, (size_t)BSZ * HID * 4, stream);
  hipMemsetAsync(hbuf0, 0, (size_t)BSZ * HID * 2, stream);
  hipMemsetAsync(hbuf1, 0, (size_t)BSZ * HID * 2, stream);

  // 4) batched encoder input GEMM: xpre = x @ W_ih_enc^T + b_ih + b_hh
  {
    dim3 grid(G4 / 64, ROWS_E / 32);
    gemm_bias_f32<EMB><<<grid, 32, 0, stream>>>(xenc, EMB, Wih_e, EMB,
                                                b_ih_enc, b_hh_enc, xpre, G4);
  }

  // 5) encoder recurrence (h double-buffered in f16)
  for (int t = 0; t < LE; ++t) {
    enc_step<<<G4 / 64, 32, 0, stream>>>(xpre, hbuf[t & 1], Whh_e, cbuf,
                                         h32, hbuf[(t + 1) & 1], encout, t);
  }

  // 6) decoder: attention + LSTM cell + attentional projection (logits deferred)
  for (int t = 0; t < LD; ++t) {
    attn_step<<<BSZ, 256, 0, stream>>>(encout, h32, demb, cellin, cat2, t);
    dec_step<<<G4 / 64, 32, 0, stream>>>(cellin, hbuf[t & 1], Wih_d, Whh_d,
                                         b_ih_dec, b_hh_dec, cbuf,
                                         h32, hbuf[(t + 1) & 1], cat2);
    proj_step<<<HID / 64, 32, 0, stream>>>(cat2, Wat, b_attn, proj, t);
  }

  // 7) batched output projection: logits[b*LD+t, :] = proj @ W_out^T + b_out
  {
    dim3 grid(VOC / 64, ROWS_D / 32);
    gemm_bias_f32<HID><<<grid, 32, 0, stream>>>(proj, HID, Wo, HID,
                                                b_out, nullptr, out, VOC);
  }
}
